// GlobalGOCorOpt_70007966924952
// MI455X (gfx1250) — matrix operand order repaired and checked by reading, not compiled, not run on hardware
//
#include <hip/hip_runtime.h>

#define DINLINE __device__ __forceinline__

constexpr int BB    = 2;
constexpr int NF    = 4096;
constexpr int CC    = 256;
constexpr int HH    = 64;
constexpr int WW    = 64;
constexpr int HW    = HH * WW;        // 4096
constexpr int MAPD  = 2 * HH - 1;     // 127
constexpr int MAPSZ = MAPD * MAPD;    // 16129
constexpr int NBINS = 10;

typedef __attribute__((ext_vector_type(16))) _Float16 v16h;
typedef __attribute__((ext_vector_type(8)))  float    v8f;

union FragU { uint4 q[2]; v16h v; };

// A fragment (16x32 f16). pl = per-lane base = rowPtr(lane) + hi*8.
// Chunks: [k + hi*8 .. +8) and [k + 16 + hi*8 .. +8)  (ISA 16-bit A layout).
DINLINE v16h load_af(const _Float16* __restrict__ pl, int k) {
  FragU u;
  u.q[0] = *reinterpret_cast<const uint4*>(pl + k);
  u.q[1] = *reinterpret_cast<const uint4*>(pl + k + 16);
  return u.v;
}
// B fragment (32x16 f16) from X[n][k] row-major. pl = colPtr(lane) + hi*8.
DINLINE v16h load_bf(const _Float16* __restrict__ pl, int k) {
  FragU u;
  u.q[0] = *reinterpret_cast<const uint4*>(pl + k);
  u.q[1] = *reinterpret_cast<const uint4*>(pl + k + 16);
  return u.v;
}

struct Ptrs { const _Float16 *a0, *a1, *b0, *b1, *b2, *b3; };
struct Frags { v16h a0, a1, b0, b1, b2, b3; };

DINLINE Frags load_set(const Ptrs& p, int k) {
  Frags f;
  f.a0 = load_af(p.a0, k);
  f.a1 = load_af(p.a1, k);
  f.b0 = load_bf(p.b0, k);
  f.b1 = load_bf(p.b1, k);
  f.b2 = load_bf(p.b2, k);
  f.b3 = load_bf(p.b3, k);
  return f;
}

DINLINE void do_wmma8(const Frags& f, v8f (&acc)[2][4]) {
  acc[0][0] = __builtin_amdgcn_wmma_f32_16x16x32_f16(false, f.a0, false, f.b0, (short)0, acc[0][0], false, false);
  acc[0][1] = __builtin_amdgcn_wmma_f32_16x16x32_f16(false, f.a0, false, f.b1, (short)0, acc[0][1], false, false);
  acc[0][2] = __builtin_amdgcn_wmma_f32_16x16x32_f16(false, f.a0, false, f.b2, (short)0, acc[0][2], false, false);
  acc[0][3] = __builtin_amdgcn_wmma_f32_16x16x32_f16(false, f.a0, false, f.b3, (short)0, acc[0][3], false, false);
  acc[1][0] = __builtin_amdgcn_wmma_f32_16x16x32_f16(false, f.a1, false, f.b0, (short)0, acc[1][0], false, false);
  acc[1][1] = __builtin_amdgcn_wmma_f32_16x16x32_f16(false, f.a1, false, f.b1, (short)0, acc[1][1], false, false);
  acc[1][2] = __builtin_amdgcn_wmma_f32_16x16x32_f16(false, f.a1, false, f.b2, (short)0, acc[1][2], false, false);
  acc[1][3] = __builtin_amdgcn_wmma_f32_16x16x32_f16(false, f.a1, false, f.b3, (short)0, acc[1][3], false, false);
}

// Double-buffered pipelined GEMM core: 32(M) x 64(N) per wave, K = KTOT (mult of 64).
template <int KTOT>
DINLINE void gemm_core(const Ptrs& p, v8f (&acc)[2][4]) {
  Frags fa = load_set(p, 0);
  Frags fb = load_set(p, 32);
  int k = 64;
#pragma unroll 2
  for (int i = 0; i < KTOT / 64 - 1; ++i) {
    do_wmma8(fa, acc);
    fa = load_set(p, k);
    k += 32;
    do_wmma8(fb, acc);
    fb = load_set(p, k);
    k += 32;
  }
  do_wmma8(fa, acc);
  do_wmma8(fb, acc);
}

// Sum across the 16 lanes of each half-wave (wave32): every lane ends with the group sum.
DINLINE float red16(float v) {
  v += __shfl_xor(v, 1, 32);
  v += __shfl_xor(v, 2, 32);
  v += __shfl_xor(v, 4, 32);
  v += __shfl_xor(v, 8, 32);
  return v;
}

// ---------------------------------------------------------------------------
// Precompute the three 127x127 maps: label_full, vplus_full, sigmoid(mask_full)
// ---------------------------------------------------------------------------
__global__ __launch_bounds__(256) void k_build_maps(
    const float* __restrict__ w_label, const float* __restrict__ w_spatial,
    const float* __restrict__ w_mask,
    float* __restrict__ labM, float* __restrict__ vpsM, float* __restrict__ wmM) {
  const int i = blockIdx.x * blockDim.x + threadIdx.x;
  if (i >= MAPSZ) return;
  const int y = i / MAPD, x = i % MAPD;
  const float d0 = (float)y - 63.0f;
  const float d1 = (float)x - 63.0f;
  const float dist = sqrtf(d0 * d0 + d1 * d1);
  float l = 0.f, v = 0.f, m = 0.f;
#pragma unroll
  for (int k = 0; k < NBINS; ++k) {
    const float bd = dist * 2.0f - (float)k;   // dist / 0.5 - k
    float w;
    if (k < NBINS - 1) w = fmaxf(1.0f - fabsf(bd), 0.0f);
    else               w = fminf(fmaxf(1.0f + bd, 0.0f), 1.0f);
    l += w_label[k] * w;
    v += w_spatial[k] * w;
    m += w_mask[k] * w;
  }
  labM[i] = l;
  vpsM[i] = v;
  wmM[i]  = 1.0f / (1.0f + expf(-m));
}

// ---------------------------------------------------------------------------
// feat (B,C,HW) f32 -> f16 in both layouts (C,HW) and transposed (HW,C)
// ---------------------------------------------------------------------------
__global__ __launch_bounds__(256) void k_convert_feat(
    const float* __restrict__ feat, _Float16* __restrict__ feat_h,
    _Float16* __restrict__ featT_h) {
  const size_t i = (size_t)blockIdx.x * blockDim.x + threadIdx.x;
  if (i >= (size_t)BB * CC * HW) return;
  const int b = (int)(i / ((size_t)CC * HW));
  const int rem = (int)(i % ((size_t)CC * HW));
  const int c = rem / HW;
  const int p = rem % HW;
  const _Float16 v = (_Float16)feat[i];
  feat_h[i] = v;
  featT_h[((size_t)b * HW + p) * CC + c] = v;
}

__global__ __launch_bounds__(256) void k_init_filt(
    const float* __restrict__ filter_map, float* __restrict__ filt,
    _Float16* __restrict__ filt_h) {
  const size_t i = (size_t)blockIdx.x * blockDim.x + threadIdx.x;
  if (i >= (size_t)BB * NF * CC) return;
  const float v = filter_map[i];
  filt[i] = v;
  filt_h[i] = (_Float16)v;
}

__global__ __launch_bounds__(256) void k_zero_alpha(float* __restrict__ a_num,
                                                    float* __restrict__ a_den) {
  const int i = blockIdx.x * blockDim.x + threadIdx.x;
  if (i < BB * NF) { a_num[i] = 0.0f; a_den[i] = 0.0f; }
}

// ---------------------------------------------------------------------------
// GEMM1 + activation epilogue:
//   scores[b,f,p] = sum_c filt[b,f,c] * feat[b,c,p]   (M=NF, N=HW, K=C)
// ---------------------------------------------------------------------------
__global__ __launch_bounds__(128) void k_scores(
    const _Float16* __restrict__ filt_h,   // [B][NF][C]
    const _Float16* __restrict__ featT_h,  // [B][HW][C]
    const float* __restrict__ labM, const float* __restrict__ vpsM,
    const float* __restrict__ wmM,
    _Float16* __restrict__ mres, _Float16* __restrict__ dact) { // [B][NF][HW]
  const int lane = threadIdx.x & 31;
  const int wave = threadIdx.x >> 5;
  const int nl = lane & 15;
  const int hi = lane >> 4;
  const int b = blockIdx.z;
  const int pBase = blockIdx.x * 64;
  const int fBase = blockIdx.y * 128 + wave * 32;

  Ptrs p;
  p.a0 = filt_h + ((size_t)b * NF + fBase + nl) * CC + hi * 8;
  p.a1 = p.a0 + (size_t)16 * CC;
  p.b0 = featT_h + (size_t)b * HW * CC + (size_t)(pBase + nl) * CC + hi * 8;
  p.b1 = p.b0 + (size_t)16 * CC;
  p.b2 = p.b0 + (size_t)32 * CC;
  p.b3 = p.b0 + (size_t)48 * CC;

  v8f acc[2][4] = {};
  gemm_core<CC>(p, acc);

#pragma unroll
  for (int mm = 0; mm < 2; ++mm) {
#pragma unroll
    for (int j = 0; j < 4; ++j) {
#pragma unroll
      for (int e = 0; e < 8; ++e) {
        const int f = fBase + mm * 16 + e + 8 * hi;
        const int pp = pBase + j * 16 + nl;
        const int iy = (HH - 1) - (f >> 6) + (pp >> 6);
        const int ix = (WW - 1) - (f & 63) + (pp & 63);
        const int mi = iy * MAPD + ix;
        const float vp = vpsM[mi];
        const float tl = labM[mi];
        const float aw = wmM[mi];
        const float s  = acc[mm][j][e];
        const float sg = (s > 0.f) ? 1.f : ((s < 0.f) ? -1.f : 0.f);
        const float act = vp * (0.5f * (1.f - aw) * fabsf(s) + 0.5f * (1.f + aw) * s);
        const float da  = vp * (0.5f * (1.f - aw) * sg + 0.5f * (1.f + aw));
        const float r   = act - vp * tl;
        const size_t oi = ((size_t)b * NF + f) * HW + pp;
        mres[oi] = (_Float16)(da * r);
        dact[oi] = (_Float16)da;
      }
    }
  }
}

// ---------------------------------------------------------------------------
// GEMM2 + epilogue:
//   fgrad[b,f,c] = rw*filt[b,f,c] + sum_p mres[b,f,p]*feat[b,c,p]   (M=NF, N=C, K=HW)
//   alpha_num[b,f] += sum_c fgrad^2
// ---------------------------------------------------------------------------
__global__ __launch_bounds__(128) void k_fgrad(
    const _Float16* __restrict__ mres,    // [B][NF][HW]
    const _Float16* __restrict__ feat_h,  // [B][C][HW]
    const float* __restrict__ filt,       // [B][NF][C]
    const float* __restrict__ filter_reg,
    float* __restrict__ fgrad, _Float16* __restrict__ fgrad_h,
    float* __restrict__ alpha_num) {
  const int lane = threadIdx.x & 31;
  const int wave = threadIdx.x >> 5;
  const int nl = lane & 15;
  const int hi = lane >> 4;
  const int b = blockIdx.z;
  const int cBase = blockIdx.x * 64;
  const int fBase = blockIdx.y * 128 + wave * 32;

  Ptrs p;
  p.a0 = mres + ((size_t)b * NF + fBase + nl) * HW + hi * 8;
  p.a1 = p.a0 + (size_t)16 * HW;
  p.b0 = feat_h + (size_t)b * CC * HW + (size_t)(cBase + nl) * HW + hi * 8;
  p.b1 = p.b0 + (size_t)16 * HW;
  p.b2 = p.b0 + (size_t)32 * HW;
  p.b3 = p.b0 + (size_t)48 * HW;

  v8f acc[2][4] = {};
  gemm_core<HW>(p, acc);

  const float fr = filter_reg[0];
  const float rw = fmaxf(fr * fr, 1e-10f);   // MIN_FILTER_REG^2
  float rowAcc[2][8] = {};
#pragma unroll
  for (int mm = 0; mm < 2; ++mm) {
#pragma unroll
    for (int j = 0; j < 4; ++j) {
#pragma unroll
      for (int e = 0; e < 8; ++e) {
        const int f = fBase + mm * 16 + e + 8 * hi;
        const int c = cBase + j * 16 + nl;
        const size_t idx = ((size_t)b * NF + f) * CC + c;
        const float g = rw * filt[idx] + acc[mm][j][e];
        fgrad[idx] = g;
        fgrad_h[idx] = (_Float16)g;
        rowAcc[mm][e] += red16(g * g);
      }
    }
  }
#pragma unroll
  for (int mm = 0; mm < 2; ++mm)
#pragma unroll
    for (int e = 0; e < 8; ++e)
      if (nl == 0)
        atomicAdd(&alpha_num[b * NF + fBase + mm * 16 + e + 8 * hi], rowAcc[mm][e]);
}

// ---------------------------------------------------------------------------
// GEMM3 + epilogue:
//   sg[b,f,p] = sum_c fgrad[b,f,c]*feat[b,c,p]    (M=NF, N=HW, K=C)
//   alpha_den[b,f] += sum_p (dact*sg)^2
// ---------------------------------------------------------------------------
__global__ __launch_bounds__(128) void k_den(
    const _Float16* __restrict__ fgrad_h,  // [B][NF][C]
    const _Float16* __restrict__ featT_h,  // [B][HW][C]
    const _Float16* __restrict__ dact,     // [B][NF][HW]
    float* __restrict__ alpha_den) {
  const int lane = threadIdx.x & 31;
  const int wave = threadIdx.x >> 5;
  const int nl = lane & 15;
  const int hi = lane >> 4;
  const int b = blockIdx.z;
  const int pBase = blockIdx.x * 64;
  const int fBase = blockIdx.y * 128 + wave * 32;

  Ptrs p;
  p.a0 = fgrad_h + ((size_t)b * NF + fBase + nl) * CC + hi * 8;
  p.a1 = p.a0 + (size_t)16 * CC;
  p.b0 = featT_h + (size_t)b * HW * CC + (size_t)(pBase + nl) * CC + hi * 8;
  p.b1 = p.b0 + (size_t)16 * CC;
  p.b2 = p.b0 + (size_t)32 * CC;
  p.b3 = p.b0 + (size_t)48 * CC;

  v8f acc[2][4] = {};
  gemm_core<CC>(p, acc);

  float rowAcc[2][8] = {};
#pragma unroll
  for (int mm = 0; mm < 2; ++mm) {
#pragma unroll
    for (int j = 0; j < 4; ++j) {
#pragma unroll
      for (int e = 0; e < 8; ++e) {
        const int f = fBase + mm * 16 + e + 8 * hi;
        const int pp = pBase + j * 16 + nl;
        const size_t oi = ((size_t)b * NF + f) * HW + pp;
        const float dv = (float)dact[oi] * acc[mm][j][e];
        rowAcc[mm][e] += red16(dv * dv);
      }
    }
  }
#pragma unroll
  for (int mm = 0; mm < 2; ++mm)
#pragma unroll
    for (int e = 0; e < 8; ++e)
      if (nl == 0)
        atomicAdd(&alpha_den[b * NF + fBase + mm * 16 + e + 8 * hi], rowAcc[mm][e]);
}

// ---------------------------------------------------------------------------
// filt -= sl * alpha * fgrad ;  alpha = num / max(den + rw*num, 1e-8)
// ---------------------------------------------------------------------------
__global__ __launch_bounds__(256) void k_update(
    float* __restrict__ filt, _Float16* __restrict__ filt_h,
    const float* __restrict__ fgrad,
    const float* __restrict__ alpha_num, const float* __restrict__ alpha_den,
    const float* __restrict__ log_sl, const float* __restrict__ filter_reg) {
  const size_t i = (size_t)blockIdx.x * blockDim.x + threadIdx.x;
  if (i >= (size_t)BB * NF * CC) return;
  const int bf = (int)(i / CC);
  const float num = alpha_num[bf];
  const float den = alpha_den[bf];
  const float fr = filter_reg[0];
  const float rw = fmaxf(fr * fr, 1e-10f);
  const float d = fmaxf(den + rw * num, 1e-8f);   // STEPLEN_REG == 0
  const float alpha = num / d;
  const float sl = expf(log_sl[0]);
  const float v = filt[i] - sl * alpha * fgrad[i];
  filt[i] = v;
  filt_h[i] = (_Float16)v;
}

// ---------------------------------------------------------------------------
extern "C" void kernel_launch(void* const* d_in, const int* in_sizes, int n_in,
                              void* d_out, int out_size, void* d_ws, size_t ws_size,
                              hipStream_t stream) {
  const float* filter_map = (const float*)d_in[0];
  const float* ref_feat   = (const float*)d_in[1];
  // d_in[2] = query_feat (unused by the reference)
  const float* w_label    = (const float*)d_in[3];
  const float* w_spatial  = (const float*)d_in[4];
  const float* w_mask     = (const float*)d_in[5];
  const float* log_sl     = (const float*)d_in[6];
  const float* freg       = (const float*)d_in[7];
  float* filt = (float*)d_out;     // working filter, (B,NF,C) f32 == final output

  char* base = (char*)d_ws;
  size_t off = 0;
  auto take = [&](size_t bytes) -> void* {
    void* p = base + off;
    off += (bytes + 255) & ~(size_t)255;
    return p;
  };
  _Float16* mres    = (_Float16*)take((size_t)BB * NF * HW * sizeof(_Float16));
  _Float16* dact    = (_Float16*)take((size_t)BB * NF * HW * sizeof(_Float16));
  _Float16* feat_h  = (_Float16*)take((size_t)BB * CC * HW * sizeof(_Float16));
  _Float16* featT_h = (_Float16*)take((size_t)BB * HW * CC * sizeof(_Float16));
  _Float16* filt_h  = (_Float16*)take((size_t)BB * NF * CC * sizeof(_Float16));
  float*    fgrad   = (float*)take((size_t)BB * NF * CC * sizeof(float));
  _Float16* fgrad_h = (_Float16*)take((size_t)BB * NF * CC * sizeof(_Float16));
  float*    a_num   = (float*)take((size_t)BB * NF * sizeof(float));
  float*    a_den   = (float*)take((size_t)BB * NF * sizeof(float));
  float*    labM    = (float*)take((size_t)MAPSZ * sizeof(float));
  float*    vpsM    = (float*)take((size_t)MAPSZ * sizeof(float));
  float*    wmM     = (float*)take((size_t)MAPSZ * sizeof(float));

  // Setup
  k_build_maps<<<(MAPSZ + 255) / 256, 256, 0, stream>>>(w_label, w_spatial, w_mask,
                                                        labM, vpsM, wmM);
  {
    const size_t n = (size_t)BB * CC * HW;
    k_convert_feat<<<(unsigned)((n + 255) / 256), 256, 0, stream>>>(ref_feat, feat_h, featT_h);
  }
  {
    const size_t n = (size_t)BB * NF * CC;
    k_init_filt<<<(unsigned)((n + 255) / 256), 256, 0, stream>>>(filter_map, filt, filt_h);
  }

  const dim3 gridBig(HW / 64, NF / 128, BB);   // 64 x 32 x 2
  const dim3 gridFg(CC / 64, NF / 128, BB);    //  4 x 32 x 2

  for (int it = 0; it < 2; ++it) {
    k_scores<<<gridBig, 128, 0, stream>>>(filt_h, featT_h, labM, vpsM, wmM, mres, dact);
    k_zero_alpha<<<(BB * NF + 255) / 256, 256, 0, stream>>>(a_num, a_den);
    k_fgrad<<<gridFg, 128, 0, stream>>>(mres, feat_h, filt, freg, fgrad, fgrad_h, a_num);
    k_den<<<gridBig, 128, 0, stream>>>(fgrad_h, featT_h, dact, a_den);
    {
      const size_t n = (size_t)BB * NF * CC;
      k_update<<<(unsigned)((n + 255) / 256), 256, 0, stream>>>(filt, filt_h, fgrad,
                                                                a_num, a_den, log_sl, freg);
    }
  }
  (void)in_sizes; (void)n_in; (void)out_size; (void)ws_size;
}
